// VectorQuantizerEMA_12927851561031
// MI455X (gfx1250) — matrix operand order repaired and checked by reading, not compiled, not run on hardware
//
#include <hip/hip_runtime.h>

// ---------------------------------------------------------------------------
// VQ-VAE vector quantizer for MI455X (gfx1250, wave32, WMMA)
// dist argmin via split-bf16 (hi/lo) GEMM on v_wmma_f32_16x16x32_bf16,
// dual accumulator chains + double-buffered async-to-LDS B-tile staging.
// ---------------------------------------------------------------------------

#define N_ROWS 131072
#define DIM    512
#define KCB    1024
#define ROWS_PER_WG 64
#define WG_THREADS  128          // 4 waves x 16 rows
#define KT_PAIRS (KCB / 32)      // 32 iterations, 32 codebook cols each
#define DSTEPS   (DIM / 32)      // 16 depth steps of 32

typedef __attribute__((ext_vector_type(16))) __bf16 v16bf;
typedef __attribute__((ext_vector_type(8)))  float  v8f;
typedef int gvec4i __attribute__((vector_size(16)));   // matches builtin V4i

#define AS1 __attribute__((address_space(1)))
#define AS3 __attribute__((address_space(3)))

#if defined(__has_builtin)
#if __has_builtin(__builtin_amdgcn_global_load_async_to_lds_b128) && \
    __has_builtin(__builtin_amdgcn_s_wait_asynccnt)
#define USE_ASYNC_LDS 1
#endif
#endif

union Frag {
    v16bf bf;
    uint4 q[2];
};

__device__ __forceinline__ unsigned short f32_to_bf16_rne(float f) {
    unsigned u = __float_as_uint(f);
    unsigned r = u + 0x7FFFu + ((u >> 16) & 1u);   // round-to-nearest-even
    return (unsigned short)(r >> 16);
}

__device__ __forceinline__ void bf16_split(float f, unsigned short& hi,
                                           unsigned short& lo) {
    hi = f32_to_bf16_rne(f);
    float fhi = __uint_as_float(((unsigned)hi) << 16);
    lo = f32_to_bf16_rne(f - fhi);
}

// ---------------------------------------------------------------------------
// Kernel 1: split codebook into bf16 hi/lo planes + compute ||w_k||^2
// ---------------------------------------------------------------------------
__global__ __launch_bounds__(128)
void vq_prep(const float* __restrict__ w, unsigned short* __restrict__ whi,
             unsigned short* __restrict__ wlo, float* __restrict__ wnorm) {
    __shared__ float red[128];
    const int k = blockIdx.x;
    const int t = threadIdx.x;
    const float4 wv = ((const float4*)(w + (size_t)k * DIM))[t];

    unsigned short h0, h1, h2, h3, l0, l1, l2, l3;
    bf16_split(wv.x, h0, l0);
    bf16_split(wv.y, h1, l1);
    bf16_split(wv.z, h2, l2);
    bf16_split(wv.w, h3, l3);

    uint2 hp, lp;
    hp.x = (unsigned)h0 | ((unsigned)h1 << 16);
    hp.y = (unsigned)h2 | ((unsigned)h3 << 16);
    lp.x = (unsigned)l0 | ((unsigned)l1 << 16);
    lp.y = (unsigned)l2 | ((unsigned)l3 << 16);
    ((uint2*)(whi + (size_t)k * DIM))[t] = hp;
    ((uint2*)(wlo + (size_t)k * DIM))[t] = lp;

    red[t] = wv.x * wv.x + wv.y * wv.y + wv.z * wv.z + wv.w * wv.w;
    __syncthreads();
    for (int off = 64; off > 0; off >>= 1) {
        if (t < off) red[t] += red[t + off];
        __syncthreads();
    }
    if (t == 0) wnorm[k] = red[0];
}

// ---------------------------------------------------------------------------
// Kernel 2: fused split-bf16 distance GEMM + running argmin.
// grid = N/64 blocks, 128 threads (4 waves x 16 rows each).
// LDS (260 KB): A hi/lo [64x512 bf16 each], B double buffer [2 x (32x512)x2],
// staged wnorm [1024 f32].
// ---------------------------------------------------------------------------
__global__ __launch_bounds__(WG_THREADS)
void vq_argmin(const float* __restrict__ x, const unsigned short* __restrict__ whi,
               const unsigned short* __restrict__ wlo,
               const float* __restrict__ wnorm, int* __restrict__ idx_out) {
    extern __shared__ unsigned short smem[];
    unsigned short* Ahi = smem;                 // [0,       32768) shorts
    unsigned short* Alo = smem + 32768;         // [32768,   65536)
    // B buffers: buf b at 65536 + b*32768; hi plane 16384 shorts, then lo.
    float* WnLds = (float*)(smem + 131072);     // 1024 floats

    const int tid  = threadIdx.x;
    const int wave = tid >> 5;
    const int lane = tid & 31;
    const int half = (lane < 16) ? 0 : 1;       // which K-half this lane owns
    const int l16  = lane & 15;
    const size_t rbase = (size_t)blockIdx.x * ROWS_PER_WG;

    // --- stage wnorm into LDS ---
    for (int i = tid; i < KCB; i += WG_THREADS) WnLds[i] = wnorm[i];

    // --- stage + convert this WG's 64 x-rows into LDS (bf16 hi/lo) ---
    const float4* xq = (const float4*)(x + rbase * DIM);
    for (int i = tid; i < (ROWS_PER_WG * DIM) / 4; i += WG_THREADS) {
        float4 v = xq[i];
        unsigned short h0, h1, h2, h3, l0, l1, l2, l3;
        bf16_split(v.x, h0, l0);
        bf16_split(v.y, h1, l1);
        bf16_split(v.z, h2, l2);
        bf16_split(v.w, h3, l3);
        uint2 hp, lp;
        hp.x = (unsigned)h0 | ((unsigned)h1 << 16);
        hp.y = (unsigned)h2 | ((unsigned)h3 << 16);
        lp.x = (unsigned)l0 | ((unsigned)l1 << 16);
        lp.y = (unsigned)l2 | ((unsigned)l3 << 16);
        ((uint2*)Ahi)[i] = hp;
        ((uint2*)Alo)[i] = lp;
    }

    // --- B-tile staging helper: 32 codebook rows (hi+lo) -> LDS buffer ---
    auto stageB = [&](int buf, int it) {
        const int n0 = it * 32;
#if defined(USE_ASYNC_LDS)
        AS1 gvec4i* gh = (AS1 gvec4i*)(whi + (size_t)n0 * DIM);
        AS1 gvec4i* gl = (AS1 gvec4i*)(wlo + (size_t)n0 * DIM);
        AS3 gvec4i* sh = (AS3 gvec4i*)(smem + 65536 + buf * 32768);
        AS3 gvec4i* sl = sh + 2048;
        for (int i = tid; i < (32 * DIM) / 8; i += WG_THREADS) {
            __builtin_amdgcn_global_load_async_to_lds_b128(gh + i, sh + i, 0, 0);
            __builtin_amdgcn_global_load_async_to_lds_b128(gl + i, sl + i, 0, 0);
        }
#else
        const uint4* gh = (const uint4*)(whi + (size_t)n0 * DIM);
        const uint4* gl = (const uint4*)(wlo + (size_t)n0 * DIM);
        uint4* sh = (uint4*)(smem + 65536 + buf * 32768);
        uint4* sl = sh + 2048;
        for (int i = tid; i < (32 * DIM) / 8; i += WG_THREADS) {
            sh[i] = gh[i];
            sl[i] = gl[i];
        }
#endif
        if (it + 2 < KT_PAIRS) {  // warm L2 for the tile after next
            __builtin_prefetch(whi + (size_t)(n0 + 64) * DIM + tid * 128, 0, 1);
            __builtin_prefetch(wlo + (size_t)(n0 + 64) * DIM + tid * 128, 0, 1);
        }
    };
#if defined(USE_ASYNC_LDS)
#define WAIT_STAGE() __builtin_amdgcn_s_wait_asynccnt(0)
#else
#define WAIT_STAGE()
#endif

    float rmin[8];
    int   ridx[8];
#pragma unroll
    for (int j = 0; j < 8; ++j) { rmin[j] = 3.0e38f; ridx[j] = 0; }

    // A layout: lane L(<16) holds row M=L, K = {0..15};
    // lane L(>=16) holds row M=L-16, K = {16..31}  -> two b128 per plane.
    const int arow = wave * 16 + l16;
    const unsigned short* aHiRow = Ahi + arow * DIM;
    const unsigned short* aLoRow = Alo + arow * DIM;

    // prologue: stage first B pair; barrier also publishes A + wnorm staging
    stageB(0, 0);
    WAIT_STAGE();
    __syncthreads();

    for (int it = 0; it < KT_PAIRS; ++it) {
        const int buf = it & 1;
        if (it + 1 < KT_PAIRS) stageB(buf ^ 1, it + 1);   // async prefetch

        const unsigned short* Bhi = smem + 65536 + buf * 32768;
        const unsigned short* Blo = Bhi + 16384;

        v8f acc0 = {0.f, 0.f, 0.f, 0.f, 0.f, 0.f, 0.f, 0.f};
        v8f acc1 = {0.f, 0.f, 0.f, 0.f, 0.f, 0.f, 0.f, 0.f};
#pragma unroll 4
        for (int t = 0; t < DSTEPS; ++t) {
            Frag ah, al, bh0, bl0, bh1, bl1;
            const int aoff = t * 32 + half * 16;
            ah.q[0] = *(const uint4*)(aHiRow + aoff);
            ah.q[1] = *(const uint4*)(aHiRow + aoff + 8);
            al.q[0] = *(const uint4*)(aLoRow + aoff);
            al.q[1] = *(const uint4*)(aLoRow + aoff + 8);
            // B layout: lane holds N=l16 (+16 for 2nd tile), K run contiguous
            const int boff0 = l16 * DIM + t * 32 + half * 16;
            const int boff1 = boff0 + 16 * DIM;
            bh0.q[0] = *(const uint4*)(Bhi + boff0);
            bh0.q[1] = *(const uint4*)(Bhi + boff0 + 8);
            bl0.q[0] = *(const uint4*)(Blo + boff0);
            bl0.q[1] = *(const uint4*)(Blo + boff0 + 8);
            bh1.q[0] = *(const uint4*)(Bhi + boff1);
            bh1.q[1] = *(const uint4*)(Bhi + boff1 + 8);
            bl1.q[0] = *(const uint4*)(Blo + boff1);
            bl1.q[1] = *(const uint4*)(Blo + boff1 + 8);
            // split-bf16: lo*hi + hi*lo + hi*hi; two independent chains
            acc0 = __builtin_amdgcn_wmma_f32_16x16x32_bf16(
                false, al.bf, false, bh0.bf, (short)0, acc0, false, false);
            acc1 = __builtin_amdgcn_wmma_f32_16x16x32_bf16(
                false, al.bf, false, bh1.bf, (short)0, acc1, false, false);
            acc0 = __builtin_amdgcn_wmma_f32_16x16x32_bf16(
                false, ah.bf, false, bl0.bf, (short)0, acc0, false, false);
            acc1 = __builtin_amdgcn_wmma_f32_16x16x32_bf16(
                false, ah.bf, false, bl1.bf, (short)0, acc1, false, false);
            acc0 = __builtin_amdgcn_wmma_f32_16x16x32_bf16(
                false, ah.bf, false, bh0.bf, (short)0, acc0, false, false);
            acc1 = __builtin_amdgcn_wmma_f32_16x16x32_bf16(
                false, ah.bf, false, bh1.bf, (short)0, acc1, false, false);
        }

        // C layout: VGPR j -> row (j + half*8); cols n0+l16 and n0+16+l16.
        const int n_a = it * 32 + l16;
        const int n_b = n_a + 16;
        const float wn_a = WnLds[n_a];
        const float wn_b = WnLds[n_b];
#pragma unroll
        for (int j = 0; j < 8; ++j) {
            float d0 = wn_a - 2.0f * acc0[j];   // ||x||^2 omitted (row-const)
            if (d0 < rmin[j]) { rmin[j] = d0; ridx[j] = n_a; }
            float d1 = wn_b - 2.0f * acc1[j];
            if (d1 < rmin[j]) { rmin[j] = d1; ridx[j] = n_b; }
        }

        WAIT_STAGE();        // next buffer landed (this wave's share)
        __syncthreads();     // all waves done reading + staging visible
    }

    // Cross-lane argmin over the 16 n-residues (halves reduce independently).
#pragma unroll
    for (int j = 0; j < 8; ++j) {
        float m = rmin[j];
        int   id = ridx[j];
        for (int off = 1; off < 16; off <<= 1) {
            float om = __shfl_xor(m, off, 32);
            int   oi = __shfl_xor(id, off, 32);
            if (om < m || (om == m && oi < id)) { m = om; id = oi; }
        }
        rmin[j] = m;
        ridx[j] = id;
    }
    if (l16 == 0) {
        const size_t row = rbase + (size_t)wave * 16 + half * 8;
#pragma unroll
        for (int j = 0; j < 8; ++j) idx_out[row + j] = ridx[j];
    }
}

// ---------------------------------------------------------------------------
// Kernel 3: gather fp32 codebook rows to output, MSE partials, histogram.
// ---------------------------------------------------------------------------
__global__ __launch_bounds__(256)
void vq_gather(const float* __restrict__ x, const float* __restrict__ w,
               const int* __restrict__ idx, float* __restrict__ out,
               int* __restrict__ counts, float* __restrict__ partial) {
    __shared__ float red[256];
    const int tid = threadIdx.x;
    const size_t row = (size_t)blockIdx.x * 2 + (tid >> 7);
    const int c4 = tid & 127;
    const int k = idx[row];

    const float4 wv = ((const float4*)(w + (size_t)k * DIM))[c4];
    const float4 xv = ((const float4*)(x + row * DIM))[c4];
    ((float4*)(out + row * DIM))[c4] = wv;   // quantized_st == quantized value

    const float dx = wv.x - xv.x, dy = wv.y - xv.y;
    const float dz = wv.z - xv.z, dw = wv.w - xv.w;
    red[tid] = dx * dx + dy * dy + dz * dz + dw * dw;
    if (c4 == 0) atomicAdd(&counts[k], 1);   // integer atomics: deterministic
    __syncthreads();
    for (int off = 128; off > 0; off >>= 1) {
        if (tid < off) red[tid] += red[tid + off];
        __syncthreads();
    }
    if (tid == 0) partial[blockIdx.x] = red[0];
}

// ---------------------------------------------------------------------------
// Kernel 4: scalar loss. 1 block, deterministic reductions.
// ---------------------------------------------------------------------------
__global__ __launch_bounds__(256)
void vq_finalize(const int* __restrict__ counts, const float* __restrict__ partial,
                 int n_partial, float* __restrict__ out_scalar) {
    __shared__ float red[256];
    const int tid = threadIdx.x;

    float msum = 0.f;
    for (int i = tid; i < n_partial; i += 256) msum += partial[i];

    const float invN = 1.0f / (float)N_ROWS;
    const float invK = 1.0f / (float)KCB;
    float reg = 0.f;
    for (int k = tid; k < KCB; k += 256) {
        const float p = (float)counts[k] * invN;
        reg += p * logf(p + 1e-10f);                 // -entropy contribution
        const float d = p - invK;
        reg += d * d;                                // diversity
        reg += p * logf(p * (float)KCB + 1e-10f);    // kl
    }

    red[tid] = msum;
    __syncthreads();
    for (int off = 128; off > 0; off >>= 1) {
        if (tid < off) red[tid] += red[tid + off];
        __syncthreads();
    }
    const float mse_sum = red[0];
    __syncthreads();
    red[tid] = reg;
    __syncthreads();
    for (int off = 128; off > 0; off >>= 1) {
        if (tid < off) red[tid] += red[tid + off];
        __syncthreads();
    }
    if (tid == 0) {
        const float mse = mse_sum / ((float)N_ROWS * (float)DIM);
        // e_latent == q_latent numerically -> loss = (1 + 0.25) * mse
        out_scalar[0] = 1.25f * mse + red[0];
    }
}

// ---------------------------------------------------------------------------
extern "C" void kernel_launch(void* const* d_in, const int* in_sizes, int n_in,
                              void* d_out, int out_size, void* d_ws, size_t ws_size,
                              hipStream_t stream) {
    (void)in_sizes; (void)n_in; (void)out_size; (void)ws_size;
    const float* x = (const float*)d_in[0];
    const float* w = (const float*)d_in[1];
    float* out = (float*)d_out;
    char* ws = (char*)d_ws;

    unsigned short* whi = (unsigned short*)(ws);             // 1 MiB
    unsigned short* wlo = (unsigned short*)(ws + 0x100000);  // 1 MiB
    float* wnorm  = (float*)(ws + 0x200000);                 // 4 KiB
    int*   idxb   = (int*)(ws + 0x201000);                   // 512 KiB
    int*   counts = (int*)(ws + 0x281000);                   // 4 KiB
    float* partial = (float*)(ws + 0x290000);                // 256 KiB

    (void)hipMemsetAsync(counts, 0, KCB * sizeof(int), stream);

    vq_prep<<<KCB, 128, 0, stream>>>(w, whi, wlo, wnorm);
    vq_argmin<<<N_ROWS / ROWS_PER_WG, WG_THREADS, 266240, stream>>>(
        x, whi, wlo, wnorm, idxb);
    vq_gather<<<N_ROWS / 2, 256, 0, stream>>>(x, w, idxb, out, counts, partial);
    vq_finalize<<<1, 256, 0, stream>>>(counts, partial, N_ROWS / 2,
                                       out + (size_t)N_ROWS * DIM);
}
